// DlrmSmall_64467459113261
// MI455X (gfx1250) — compile-verified, matrix operand order checked by reading, not run on hardware
//
#include <hip/hip_runtime.h>
#include <hip/hip_bf16.h>

// ---------------------------------------------------------------------------
// DLRM-small forward, bf16 WMMA (CDNA5 / gfx1250, wave32)
// ---------------------------------------------------------------------------

typedef __bf16 bf16_t;
typedef bf16_t bf16x16 __attribute__((ext_vector_type(16)));
typedef float  f32x8   __attribute__((ext_vector_type(8)));
typedef float  f32x4   __attribute__((ext_vector_type(4)));
typedef unsigned int u32;
typedef u32 u32x2 __attribute__((ext_vector_type(2)));
typedef u32 u32x4 __attribute__((ext_vector_type(4)));

#define DLRM_B     16384
#define DLRM_VOCAB 50000

static __device__ __forceinline__ f32x8 wmma_bf16(bf16x16 a, bf16x16 b, f32x8 c) {
    return __builtin_amdgcn_wmma_f32_16x16x32_bf16(false, a, false, b, (short)0, c,
                                                   false, false);
}

// ---------------------------------------------------------------------------
// Weight convert: f32 row-major [K x N] -> bf16 transposed [N x Kpad] (k>=K -> 0)
// ---------------------------------------------------------------------------
__global__ void convert_wt_kernel(const float* __restrict__ src, bf16_t* __restrict__ dst,
                                  int K, int N, int Kpad) {
    int i = blockIdx.x * 256 + threadIdx.x;
    if (i >= N * Kpad) return;
    int n = i / Kpad;
    int k = i - n * Kpad;
    float v = (k < K) ? src[(size_t)k * N + n] : 0.0f;
    dst[i] = (bf16_t)v;
}

// dense [B,13] f32 -> [B,32] bf16, padded with zeros
__global__ void convert_dense_kernel(const float* __restrict__ src, bf16_t* __restrict__ dst) {
    int i = blockIdx.x * 256 + threadIdx.x;   // B*32 threads
    int b = i >> 5;
    int k = i & 31;
    float v = (k < 13) ? src[(size_t)b * 13 + k] : 0.0f;
    dst[i] = (bf16_t)v;
}

// ---------------------------------------------------------------------------
// Tiled bf16 GEMM:  O[M,N] = relu(A[M,K] * Wt[N,K]^T + bias)
// block = 256 threads (8 waves), tile 128x128, K-step 32.
// Ping-pong double-buffered LDS: one barrier per K-step; global loads for
// tile k+1 overlap the 8 WMMAs of tile k.
// ---------------------------------------------------------------------------
#define GEMM_LDK 40   // 32 + 8 pad (80B row => 16B aligned, staggered banks)

__global__ __launch_bounds__(256)
void gemm_bf16_kernel(const bf16_t* __restrict__ A, const bf16_t* __restrict__ Wt,
                      const float* __restrict__ bias, bf16_t* __restrict__ O,
                      int K, int N, int do_relu) {
    __shared__ bf16_t sA[2][128 * GEMM_LDK];
    __shared__ bf16_t sB[2][128 * GEMM_LDK];

    const int tid   = threadIdx.x;
    const int lane  = tid & 31;
    const int wv    = tid >> 5;
    const int waveM = wv & 3;      // 0..3
    const int waveN = wv >> 2;     // 0..1
    const int lm    = lane & 15;
    const int lh    = lane >> 4;

    const int m0 = blockIdx.x * 128;
    const int n0 = blockIdx.y * 128;

    const int ldRow = tid >> 1;            // 0..127
    const int ldCol = (tid & 1) * 16;      // 0 or 16

    const bf16_t* gA = A  + (size_t)(m0 + ldRow) * K + ldCol;
    const bf16_t* gB = Wt + (size_t)(n0 + ldRow) * K + ldCol;
    const int sOff  = ldRow * GEMM_LDK + ldCol;

    f32x8 acc[2][4];
    for (int mi = 0; mi < 2; ++mi)
        for (int ni = 0; ni < 4; ++ni)
            acc[mi][ni] = f32x8{};

    // prologue: stage tile 0 into buffer 0
    {
        u32x4 va0 = ((const u32x4*)gA)[0];
        u32x4 va1 = ((const u32x4*)gA)[1];
        u32x4 vb0 = ((const u32x4*)gB)[0];
        u32x4 vb1 = ((const u32x4*)gB)[1];
        *((u32x4*)&sA[0][sOff])     = va0;
        *((u32x4*)&sA[0][sOff + 8]) = va1;
        *((u32x4*)&sB[0][sOff])     = vb0;
        *((u32x4*)&sB[0][sOff + 8]) = vb1;
    }
    __syncthreads();

    int cur = 0;
    for (int k0 = 0; k0 < K; k0 += 32) {
        const bool more = (k0 + 32) < K;

        // issue global loads for the NEXT tile before computing this one
        u32x4 va0, va1, vb0, vb1;
        if (more) {
            const bf16_t* ga = gA + k0 + 32;
            const bf16_t* gb = gB + k0 + 32;
            if (k0 + 64 < K) {   // gfx1250 global_prefetch_b8 two tiles ahead
                __builtin_prefetch((const void*)(ga + 32), 0, 1);
                __builtin_prefetch((const void*)(gb + 32), 0, 1);
            }
            va0 = ((const u32x4*)ga)[0];
            va1 = ((const u32x4*)ga)[1];
            vb0 = ((const u32x4*)gb)[0];
            vb1 = ((const u32x4*)gb)[1];
        }

        // compute tile k from buffer `cur`
        bf16x16 aF[2], bF[4];
        for (int mi = 0; mi < 2; ++mi)
            aF[mi] = *((const bf16x16*)&sA[cur][(waveM * 32 + mi * 16 + lm) * GEMM_LDK + lh * 16]);
        for (int ni = 0; ni < 4; ++ni)
            bF[ni] = *((const bf16x16*)&sB[cur][(waveN * 64 + ni * 16 + lm) * GEMM_LDK + lh * 16]);
        for (int mi = 0; mi < 2; ++mi)
            for (int ni = 0; ni < 4; ++ni)
                acc[mi][ni] = wmma_bf16(aF[mi], bF[ni], acc[mi][ni]);

        // stage tile k+1 into buffer `nxt`; single barrier per iteration
        if (more) {
            const int nxt = cur ^ 1;
            *((u32x4*)&sA[nxt][sOff])     = va0;
            *((u32x4*)&sA[nxt][sOff + 8]) = va1;
            *((u32x4*)&sB[nxt][sOff])     = vb0;
            *((u32x4*)&sB[nxt][sOff + 8]) = vb1;
            __syncthreads();
            cur = nxt;
        }
    }

    // Epilogue: bias + ReLU + bf16 store. C layout: VGPR r -> M = r + 8*(lane/16)
    for (int mi = 0; mi < 2; ++mi) {
        for (int ni = 0; ni < 4; ++ni) {
            const int colg = n0 + waveN * 64 + ni * 16 + lm;
            const float bv = bias[colg];
            for (int r = 0; r < 8; ++r) {
                const int rowg = m0 + waveM * 32 + mi * 16 + r + lh * 8;
                float v = acc[mi][ni][r] + bv;
                if (do_relu) v = v > 0.0f ? v : 0.0f;
                O[(size_t)rowg * N + colg] = (bf16_t)v;
            }
        }
    }
}

// ---------------------------------------------------------------------------
// Fused embedding gather + dot-interaction.
// One wave per sample: gather 27 feature rows (1 bottom-MLP + 26 embeddings)
// into LDS as bf16 [32 x 136] (rows 27..31 zero), compute S = F*F^T via WMMA
// (Gram => A and B fragments are the same contiguous per-row LDS reads),
// emit upper-tri(378) + x(128) + zero-pad(6) into topin [B, 512] bf16.
// ---------------------------------------------------------------------------
#define FLD 136   // 128 + 8 pad; 272B rows => 16B aligned, staggered banks

__global__ __launch_bounds__(128)
void interact_kernel(const bf16_t* __restrict__ X3, const int* __restrict__ cat,
                     const float* __restrict__ emb, bf16_t* __restrict__ topin) {
    __shared__ bf16_t sF[4 * 32 * FLD];   // 4 waves/block, 1 sample/wave (34.8 KB)

    const int tid  = threadIdx.x;
    const int lane = tid & 31;
    const int wv   = tid >> 5;
    const int b    = blockIdx.x * 4 + wv;
    bf16_t* F = &sF[wv * 32 * FLD];

    // one coalesced index load per lane; broadcast below via shfl
    int myIdx = 0;
    if (lane < 26) myIdx = cat[(size_t)b * 26 + lane] + lane * DLRM_VOCAB;

    // row 0: bottom-MLP output (already bf16): 8B per lane
    {
        const bf16_t* src = X3 + (size_t)b * 128 + lane * 4;
        *((u32x2*)&F[lane * 4]) = *((const u32x2*)src);
    }
    // rows 1..26: gather embedding rows (f32 -> bf16), 16B load / 8B LDS store per lane
    for (int t = 0; t < 26; ++t) {
        const int idx = __shfl(myIdx, t, 32);
        f32x4 v = *((const f32x4*)(emb + (size_t)idx * 128) + lane);
        alignas(8) bf16_t h[4];
        h[0] = (bf16_t)v[0]; h[1] = (bf16_t)v[1];
        h[2] = (bf16_t)v[2]; h[3] = (bf16_t)v[3];
        *((u32x2*)&F[(t + 1) * FLD + lane * 4]) = *((const u32x2*)h);
    }
    // rows 27..31: zero pad
    for (int r = 27; r < 32; ++r)
        *((u32x2*)&F[r * FLD + lane * 4]) = u32x2{0u, 0u};

    __syncthreads();

    const int lm = lane & 15;
    const int lh = lane >> 4;

    // needed 16x16 output tiles of the 32x32 Gram: (0,0),(0,1),(1,1)
    const int tmArr[3] = {0, 0, 1};
    const int tnArr[3] = {0, 1, 1};
    for (int tp = 0; tp < 3; ++tp) {
        const int tm = tmArr[tp], tn = tnArr[tp];
        f32x8 c = f32x8{};
        for (int ks = 0; ks < 4; ++ks) {   // K = 128 = 4 * 32
            bf16x16 a  = *((const bf16x16*)&F[(tm * 16 + lm) * FLD + ks * 32 + lh * 16]);
            bf16x16 bb = *((const bf16x16*)&F[(tn * 16 + lm) * FLD + ks * 32 + lh * 16]);
            c = wmma_bf16(a, bb, c);
        }
        for (int r = 0; r < 8; ++r) {
            const int M = tm * 16 + r + lh * 8;
            const int N = tn * 16 + lm;
            if (M <= N && N < 27) {
                const int tri = M * 27 - (M * (M - 1)) / 2 + (N - M);
                topin[(size_t)b * 512 + 128 + tri] = (bf16_t)c[r];
            }
        }
    }

    // topin[:,0:128] = x (bottom-MLP output)
    *((u32x2*)&topin[(size_t)b * 512 + lane * 4]) = *((const u32x2*)&F[lane * 4]);
    // topin[:,506:512] = 0
    if (lane < 6) topin[(size_t)b * 512 + 506 + lane] = (bf16_t)0.0f;
}

// ---------------------------------------------------------------------------
// Final layer: logits[B] = T4[B,256](bf16) . w[256] + bias   (f32 output)
// One wave per sample, 8 samples per block.
// ---------------------------------------------------------------------------
__global__ __launch_bounds__(256)
void final_layer_kernel(const bf16_t* __restrict__ T4, const float* __restrict__ w,
                        const float* __restrict__ bias, float* __restrict__ out) {
    const int tid  = threadIdx.x;
    const int lane = tid & 31;
    const int wv   = tid >> 5;
    const int b    = blockIdx.x * 8 + wv;
    float s = 0.0f;
    for (int j = lane; j < 256; j += 32)
        s += (float)T4[(size_t)b * 256 + j] * w[j];
    for (int off = 16; off > 0; off >>= 1)
        s += __shfl_xor(s, off, 32);
    if (lane == 0) out[b] = s + bias[0];
}

// ---------------------------------------------------------------------------
// Launch
// ---------------------------------------------------------------------------
extern "C" void kernel_launch(void* const* d_in, const int* in_sizes, int n_in,
                              void* d_out, int out_size, void* d_ws, size_t ws_size,
                              hipStream_t stream) {
    (void)in_sizes; (void)n_in; (void)out_size; (void)ws_size;
    const int B = DLRM_B;

    // setup_inputs() dict order, lists flattened:
    const float* dense    = (const float*)d_in[0];   // [B,13]
    const int*   cat      = (const int*)  d_in[1];   // [B,26]
    const float* bot_w0   = (const float*)d_in[2];   // [13,512]
    const float* bot_w1   = (const float*)d_in[3];   // [512,256]
    const float* bot_w2   = (const float*)d_in[4];   // [256,128]
    const float* bot_b0   = (const float*)d_in[5];
    const float* bot_b1   = (const float*)d_in[6];
    const float* bot_b2   = (const float*)d_in[7];
    const float* emb      = (const float*)d_in[8];   // [1.3M,128]
    const float* top_w0   = (const float*)d_in[9];   // [506,1024]
    const float* top_w1   = (const float*)d_in[10];  // [1024,1024]
    const float* top_w2   = (const float*)d_in[11];  // [1024,512]
    const float* top_w3   = (const float*)d_in[12];  // [512,256]
    const float* top_w4   = (const float*)d_in[13];  // [256,1]
    const float* top_b0   = (const float*)d_in[14];
    const float* top_b1   = (const float*)d_in[15];
    const float* top_b2   = (const float*)d_in[16];
    const float* top_b3   = (const float*)d_in[17];
    const float* top_b4   = (const float*)d_in[18];

    char* ws = (char*)d_ws;
    size_t off = 0;
    auto alloc = [&](size_t bytes) -> void* {
        void* p = ws + off;
        off = (off + bytes + 255) & ~(size_t)255;
        return p;
    };

    bf16_t* wB0 = (bf16_t*)alloc((size_t)512  * 32   * 2);
    bf16_t* wB1 = (bf16_t*)alloc((size_t)256  * 512  * 2);
    bf16_t* wB2 = (bf16_t*)alloc((size_t)128  * 256  * 2);
    bf16_t* wT0 = (bf16_t*)alloc((size_t)1024 * 512  * 2);
    bf16_t* wT1 = (bf16_t*)alloc((size_t)1024 * 1024 * 2);
    bf16_t* wT2 = (bf16_t*)alloc((size_t)512  * 1024 * 2);
    bf16_t* wT3 = (bf16_t*)alloc((size_t)256  * 512  * 2);
    bf16_t* A0  = (bf16_t*)alloc((size_t)B * 32   * 2);
    bf16_t* P   = (bf16_t*)alloc((size_t)B * 1024 * 2);   // ping
    bf16_t* Q   = (bf16_t*)alloc((size_t)B * 1024 * 2);   // pong
    bf16_t* X3  = (bf16_t*)alloc((size_t)B * 128  * 2);   // bottom-MLP out

    auto cvtw = [&](const float* src, bf16_t* dst, int K, int N, int Kpad) {
        int total = N * Kpad;
        convert_wt_kernel<<<(total + 255) / 256, 256, 0, stream>>>(src, dst, K, N, Kpad);
    };
    cvtw(bot_w0, wB0, 13,   512,  32);
    cvtw(bot_w1, wB1, 512,  256,  512);
    cvtw(bot_w2, wB2, 256,  128,  256);
    cvtw(top_w0, wT0, 506,  1024, 512);
    cvtw(top_w1, wT1, 1024, 1024, 1024);
    cvtw(top_w2, wT2, 1024, 512,  1024);
    cvtw(top_w3, wT3, 512,  256,  512);

    convert_dense_kernel<<<(B * 32) / 256, 256, 0, stream>>>(dense, A0);

    auto gemm = [&](const bf16_t* A, const bf16_t* Wt, const float* bias, bf16_t* O,
                    int K, int N, int relu) {
        dim3 grid(B / 128, N / 128);
        gemm_bf16_kernel<<<grid, 256, 0, stream>>>(A, Wt, bias, O, K, N, relu);
    };

    // bottom MLP: 13(pad32)->512->256->128
    gemm(A0, wB0, bot_b0, P,  32,  512, 1);   // X1
    gemm(P,  wB1, bot_b1, Q,  512, 256, 1);   // X2
    gemm(Q,  wB2, bot_b2, X3, 256, 128, 1);   // X3 (bottom out)

    // fused gather + interaction -> top input [B,512] bf16 (506 + 6 zero pad)
    interact_kernel<<<B / 4, 128, 0, stream>>>(X3, cat, emb, P);

    // top MLP: 506(pad512)->1024->1024->512->256->1
    gemm(P, wT0, top_b0, Q, 512,  1024, 1);   // T1
    gemm(Q, wT1, top_b1, P, 1024, 1024, 1);   // T2
    gemm(P, wT2, top_b2, Q, 1024, 512,  1);   // T3
    gemm(Q, wT3, top_b3, P, 512,  256,  1);   // T4

    final_layer_kernel<<<B / 8, 256, 0, stream>>>(P, top_w4, top_b4, (float*)d_out);
}
